// DPC_3745211482228
// MI455X (gfx1250) — matrix operand order repaired and checked by reading, not compiled, not run on hardware
//
#include <hip/hip_runtime.h>
#include <stdint.h>

// ---------------- WMMA types & helpers (gfx1250, wave32) ----------------
typedef __attribute__((ext_vector_type(16))) __bf16 v16bf;
typedef __attribute__((ext_vector_type(8)))  float  v8f;

union FragAB {
  v16bf v;
  uint4 q[2];
};

__device__ __forceinline__ uint16_t f2bf(float f) {
  uint32_t u = __float_as_uint(f);
  u += 0x7fffu + ((u >> 16) & 1u);   // round to nearest even
  return (uint16_t)(u >> 16);
}

// A fragment: 16x32 bf16 tile, source row-major [M][lda], tile (mt, kt).
// Layout: lanes 0-15 hold row M=lane, K = 0..15;
// lanes 16-31 hold row M=lane-16, K = 16..31.  Two 16B contiguous loads per lane.
__device__ __forceinline__ v16bf load_a(const uint16_t* A, int lda, int mt, int kt, int lane) {
  const int row = mt * 16 + (lane & 15);
  const int kh8 = (lane >> 4) << 4;                 // 0 or 16
  const uint16_t* p = A + (size_t)row * lda + kt * 32 + kh8;
  FragAB f;
  f.q[0] = *reinterpret_cast<const uint4*>(p);      // K = kh8 .. kh8+7
  f.q[1] = *reinterpret_cast<const uint4*>(p + 8);  // K = kh8+8 .. kh8+15
  return f.v;
}

// B fragment: 32x16 bf16 tile from pre-packed weights.
// Packed so each lane reads its 16 values (32 bytes) contiguously.
__device__ __forceinline__ v16bf load_b(const uint16_t* P, int ntiles, int kt, int nt, int lane) {
  const uint16_t* p = P + ((size_t)(kt * ntiles + nt) << 9) + lane * 16;
  FragAB f;
  f.q[0] = *reinterpret_cast<const uint4*>(p);
  f.q[1] = *reinterpret_cast<const uint4*>(p + 8);
  return f.v;
}

__device__ __forceinline__ v8f wmma_bf16(v16bf a, v16bf b, v8f c) {
  return __builtin_amdgcn_wmma_f32_16x16x32_bf16(false, a, false, b, (short)0, c, false, false);
}

// ---------------- problem constants ----------------
#define NB    128      // batch
#define NT    256      // time steps
#define DIN   784      // input dim
#define DINP  800      // input dim padded to mult of 32 (K for W_ih)
#define LOW   256      // lower dim
#define HIGH  64       // higher dim
#define NH1   128      // hidden 1
#define NWG   32
#define NTHR  256
#define TOTW  (NWG * 8)          // total waves
#define OUT_PE    1280           // out offsets: logits [128*10], pe, gw_last [640], rts
#define OUT_GW    1281
#define OUT_RTS   1921

// ---------------- prep kernels ----------------
// Zero bf16 state span (rt, err, rh contiguous) + barrier counter.
__global__ void dpc_init_ws(uint16_t* zspan, int n, int* bar) {
  int i = blockIdx.x * blockDim.x + threadIdx.x;
  if (i < n) zspan[i] = 0;
  if (i < 16) bar[i] = 0;
}

// Pack virtual weight W[k][n] = src[n*ldsrc + k] (zero-padded to K x N) into
// WMMA-B fragment order: tile(kt,nt) -> 512 bf16, element (lane*16 + 2p+e)
// holds W[kt*32 + (lane>>4)*16 + 2p+e][nt*16 + (lane&15)].
__global__ void dpc_pack_w(const float* __restrict__ src, uint16_t* __restrict__ dst,
                           int K, int N, int Ksrc, int Nsrc, int ldsrc) {
  int total = K * N;
  int idx = blockIdx.x * blockDim.x + threadIdx.x;
  if (idx >= total) return;
  int tile   = idx >> 9;
  int within = idx & 511;
  int lane   = within >> 4;
  int pe     = within & 15;
  int ntiles = N >> 4;
  int nt = tile % ntiles;
  int kt = tile / ntiles;
  int k = kt * 32 + ((lane >> 4) << 4) + pe;
  int n = nt * 16 + (lane & 15);
  float w = (k < Ksrc && n < Nsrc) ? src[(size_t)n * ldsrc + k] : 0.f;
  dst[idx] = f2bf(w);
}

// ---------------- device-wide barrier (persistent kernel) ----------------
__device__ __forceinline__ void global_barrier(int* cnt, int target) {
  __threadfence();                       // release all prior global writes
  __syncthreads();
  if (threadIdx.x == 0) {
    __hip_atomic_fetch_add(cnt, 1, __ATOMIC_ACQ_REL, __HIP_MEMORY_SCOPE_AGENT);
    while (__hip_atomic_load(cnt, __ATOMIC_ACQUIRE, __HIP_MEMORY_SCOPE_AGENT) < target) {
      __builtin_amdgcn_s_sleep(1);
    }
  }
  __syncthreads();
  __threadfence();                       // acquire side
}

// ---------------- main persistent step kernel ----------------
__global__ __launch_bounds__(NTHR)
void dpc_step_kernel(
    const float* __restrict__ inseq,   // [128][256][784]
    const float* __restrict__ noise,   // [256][128][256]
    const float* __restrict__ b_h1, const float* __restrict__ b_h2,
    const float* __restrict__ b_ih, const float* __restrict__ b_hh,
    const float* __restrict__ Wcls,    // [10][256] fp32 (used directly)
    const float* __restrict__ b_cls,
    uint16_t* __restrict__ rt_bf,      // [128][256]
    uint16_t* __restrict__ err_bf,     // [128][800] (cols 784.. stay zero)
    uint16_t* __restrict__ rh_bf,      // [128][64]
    uint16_t* __restrict__ h_bf,       // [128][128]
    float* __restrict__ gw_f,          // [128][16]
    float* __restrict__ vr_f,          // [5][128][256]
    const uint16_t* __restrict__ Wl_p,   // K=256 N=784
    const uint16_t* __restrict__ Wih_p,  // K=800 N=64
    const uint16_t* __restrict__ Whh_p,  // K=64  N=64
    const uint16_t* __restrict__ Wh1_p,  // K=64  N=128
    const uint16_t* __restrict__ Wh2_p,  // K=128 N=16
    const uint16_t* __restrict__ Wv_p,   // 5 x (K=256 N=256)
    int* __restrict__ bar, float* __restrict__ pe_part,
    float* __restrict__ out) {

  const int tid  = threadIdx.x;
  const int lane = tid & 31;
  const int wave = tid >> 5;
  const int wg   = blockIdx.x;
  const int gwv  = wg * 8 + wave;        // global wave id 0..255
  const int row8 = (lane >> 4) << 3;     // C/D fragment row base
  const int ccol = lane & 15;            // C/D fragment column

  __shared__ float rtrows[4][LOW];       // this WG's 4 rt_new rows (fp32)
  __shared__ float sred[8];

  int epoch = 0;
  float pe_reg = 0.f;                    // per-thread pe partial (fixed ownership)
  float logit_acc = 0.f;                 // per-thread logit partial (tid<40 only)

  for (int t = 0; t < NT; ++t) {
    // ---------- Phase A: frame error = x_t - rt @ W_lower^T (all WGs) ----------
    for (int task = gwv; task < 8 * 49; task += TOTW) {
      const int mt = task / 49, nt = task % 49;
      __builtin_prefetch(&inseq[(size_t)(mt * 16 + row8) * (NT * DIN) + (size_t)t * DIN + nt * 16], 0, 1);
      v8f acc = {0.f,0.f,0.f,0.f,0.f,0.f,0.f,0.f};
      for (int kt = 0; kt < 8; ++kt)
        acc = wmma_bf16(load_a(rt_bf, LOW, mt, kt, lane), load_b(Wl_p, 49, kt, nt, lane), acc);
      for (int r = 0; r < 8; ++r) {
        const int m = mt * 16 + row8 + r;
        const int n = nt * 16 + ccol;
        float e = inseq[(size_t)m * (NT * DIN) + (size_t)t * DIN + n] - acc[r];
        pe_reg += e * e;
        err_bf[m * DINP + n] = f2bf(e);
      }
    }
    global_barrier(bar, ++epoch * NWG);

    // ---------- Phase B: WG0 runs serial chain; WG1..31 compute vr ----------
    if (wg == 0) {
      // rh_new = relu(err @ Wih^T + rh @ Whh^T + b_ih + b_hh)  [128x64], 32 tiles
      uint16_t rhout[4][8];
      for (int i = 0; i < 4; ++i) {
        const int tile = wave + 8 * i;
        const int mt = tile >> 2, nt = tile & 3;
        v8f acc = {0.f,0.f,0.f,0.f,0.f,0.f,0.f,0.f};
        for (int kt = 0; kt < 25; ++kt)
          acc = wmma_bf16(load_a(err_bf, DINP, mt, kt, lane), load_b(Wih_p, 4, kt, nt, lane), acc);
        for (int kt = 0; kt < 2; ++kt)
          acc = wmma_bf16(load_a(rh_bf, HIGH, mt, kt, lane), load_b(Whh_p, 4, kt, nt, lane), acc);
        for (int r = 0; r < 8; ++r) {
          const int n = nt * 16 + ccol;
          rhout[i][r] = f2bf(fmaxf(acc[r] + b_ih[n] + b_hh[n], 0.f));
        }
      }
      __syncthreads();                             // all reads of old rh done
      for (int i = 0; i < 4; ++i) {
        const int tile = wave + 8 * i;
        const int mt = tile >> 2, nt = tile & 3;
        for (int r = 0; r < 8; ++r)
          rh_bf[(mt * 16 + row8 + r) * HIGH + nt * 16 + ccol] = rhout[i][r];
      }
      __threadfence_block();
      __syncthreads();
      // h = relu(rh_new @ Wh1^T + b_h1)  [128x128], 64 tiles (no WAR hazard)
      for (int i = 0; i < 8; ++i) {
        const int tile = wave + 8 * i;
        const int mt = tile >> 3, nt = tile & 7;
        v8f acc = {0.f,0.f,0.f,0.f,0.f,0.f,0.f,0.f};
        for (int kt = 0; kt < 2; ++kt)
          acc = wmma_bf16(load_a(rh_bf, HIGH, mt, kt, lane), load_b(Wh1_p, 8, kt, nt, lane), acc);
        for (int r = 0; r < 8; ++r) {
          const int n = nt * 16 + ccol;
          h_bf[(mt * 16 + row8 + r) * NH1 + n] = f2bf(fmaxf(acc[r] + b_h1[n], 0.f));
        }
      }
      __threadfence_block();
      __syncthreads();
      // gw = h @ Wh2^T + b_h2  [128x16] (cols 5..15 are zero-padded weights)
      {
        const int mt = wave;
        v8f acc = {0.f,0.f,0.f,0.f,0.f,0.f,0.f,0.f};
        for (int kt = 0; kt < 4; ++kt)
          acc = wmma_bf16(load_a(h_bf, NH1, mt, kt, lane), load_b(Wh2_p, 1, kt, 0, lane), acc);
        for (int r = 0; r < 8; ++r) {
          float v = acc[r] + (ccol < 5 ? b_h2[ccol] : 0.f);
          gw_f[(mt * 16 + row8 + r) * 16 + ccol] = v;
        }
      }
    } else {
      // vr[k] = rt @ Vk[k]^T, 5 x [128x256] = 640 tiles over 248 waves
      const int vw = (wg - 1) * 8 + wave;
      for (int task = vw; task < 640; task += 248) {
        const int kk = task >> 7;
        const int rem = task & 127;
        const int mt = rem >> 4, nt = rem & 15;
        const uint16_t* Wp = Wv_p + (size_t)kk * (LOW * LOW);
        v8f acc = {0.f,0.f,0.f,0.f,0.f,0.f,0.f,0.f};
        for (int kt = 0; kt < 8; ++kt)
          acc = wmma_bf16(load_a(rt_bf, LOW, mt, kt, lane), load_b(Wp, 16, kt, nt, lane), acc);
        for (int r = 0; r < 8; ++r)
          vr_f[(size_t)kk * (NB * LOW) + (mt * 16 + row8 + r) * LOW + nt * 16 + ccol] = acc[r];
      }
    }
    global_barrier(bar, ++epoch * NWG);

    // ---------- Phase C: rt_new, rts output, logit accumulation ----------
    {
      const int b0 = wg * 4;                       // 4 batch rows per WG
      for (int s = 0; s < 4; ++s) {
        const int j  = tid + s * NTHR;             // 0..1023
        const int bl = j >> 8;
        const int i  = j & 255;
        const int b  = b0 + bl;
        float v = 0.01f * noise[(size_t)t * (NB * LOW) + (size_t)b * LOW + i];
        for (int k = 0; k < 5; ++k)
          v += gw_f[b * 16 + k] * vr_f[(size_t)k * (NB * LOW) + (size_t)b * LOW + i];
        const float r = fmaxf(v, 0.f);
        out[OUT_RTS + (size_t)t * (NB * LOW) + (size_t)b * LOW + i] = r;
        rt_bf[b * LOW + i] = f2bf(r);
        rtrows[bl][i] = r;
      }
      __syncthreads();
      if (tid < 40) {                              // 4 rows x 10 classes
        const int bl = tid / 10, c = tid % 10;
        float s = 0.f;
        for (int i = 0; i < LOW; ++i)
          s += rtrows[bl][i] * Wcls[c * LOW + i];
        logit_acc += s;                            // fixed (b,c) owner -> deterministic
      }
    }
    global_barrier(bar, ++epoch * NWG);
  }

  // ---------- Finalize ----------
  {  // deterministic pe reduction: wave shuffle -> LDS -> per-WG partial
    float p = pe_reg;
    for (int o = 16; o > 0; o >>= 1) p += __shfl_down(p, o, 32);
    if (lane == 0) sred[wave] = p;
    __syncthreads();
    if (tid == 0) {
      float s = 0.f;
      for (int w = 0; w < 8; ++w) s += sred[w];
      pe_part[wg] = s;
    }
  }
  if (tid < 40) {                                  // digit_logit = mean over T + bias
    const int bl = tid / 10, c = tid % 10;
    out[(wg * 4 + bl) * 10 + c] = logit_acc * (1.f / (float)NT) + b_cls[c];
  }
  global_barrier(bar, ++epoch * NWG);
  if (wg == 0) {
    if (tid == 0) {
      float s = 0.f;
      for (int i = 0; i < NWG; ++i) s += pe_part[i];
      out[OUT_PE] = s / ((float)NT * (float)NB * (float)DIN);
    }
    for (int idx = tid; idx < NB * 5; idx += NTHR) {  // gw at t = T-1
      out[OUT_GW + idx] = gw_f[(idx / 5) * 16 + (idx % 5)];
    }
  }
}

// ---------------- host: workspace layout + launches ----------------
extern "C" void kernel_launch(void* const* d_in, const int* in_sizes, int n_in,
                              void* d_out, int out_size, void* d_ws, size_t ws_size,
                              hipStream_t stream) {
  (void)in_sizes; (void)n_in; (void)out_size; (void)ws_size;
  const float* inseq   = (const float*)d_in[0];
  const float* noise   = (const float*)d_in[1];
  const float* W_lower = (const float*)d_in[2];
  const float* Vk      = (const float*)d_in[3];
  const float* W_h1    = (const float*)d_in[4];
  const float* b_h1    = (const float*)d_in[5];
  const float* W_h2    = (const float*)d_in[6];
  const float* b_h2    = (const float*)d_in[7];
  const float* W_ih    = (const float*)d_in[8];
  const float* W_hh    = (const float*)d_in[9];
  const float* b_ih    = (const float*)d_in[10];
  const float* b_hh    = (const float*)d_in[11];
  const float* W_cls   = (const float*)d_in[12];
  const float* b_cls   = (const float*)d_in[13];
  float* out = (float*)d_out;

  char* ws = (char*)d_ws;
  size_t cur = 0;
  auto alloc = [&](size_t bytes) -> void* {
    void* p = ws + cur;
    cur = (cur + bytes + 255) & ~(size_t)255;
    return p;
  };
  int*      bar     = (int*)alloc(64);
  float*    pe_part = (float*)alloc(NWG * sizeof(float));
  float*    gw_f    = (float*)alloc(NB * 16 * sizeof(float));
  float*    vr_f    = (float*)alloc((size_t)5 * NB * LOW * sizeof(float));
  uint16_t* rt_bf   = (uint16_t*)alloc((size_t)NB * LOW * 2);     // |
  uint16_t* err_bf  = (uint16_t*)alloc((size_t)NB * DINP * 2);    // | contiguous zero span
  uint16_t* rh_bf   = (uint16_t*)alloc((size_t)NB * HIGH * 2);    // |
  uint16_t* h_bf    = (uint16_t*)alloc((size_t)NB * NH1 * 2);
  uint16_t* Wl_p    = (uint16_t*)alloc((size_t)256 * 784 * 2);
  uint16_t* Wih_p   = (uint16_t*)alloc((size_t)800 * 64 * 2);
  uint16_t* Whh_p   = (uint16_t*)alloc((size_t)64 * 64 * 2);
  uint16_t* Wh1_p   = (uint16_t*)alloc((size_t)64 * 128 * 2);
  uint16_t* Wh2_p   = (uint16_t*)alloc((size_t)128 * 16 * 2);
  uint16_t* Wv_p    = (uint16_t*)alloc((size_t)5 * 256 * 256 * 2);

  const int zspan = NB * LOW + NB * DINP + NB * HIGH;   // rt + err + rh (all 256B-multiples)
  dpc_init_ws<<<(zspan + 255) / 256, 256, 0, stream>>>(rt_bf, zspan, bar);

  dpc_pack_w<<<(256 * 784) / 256, 256, 0, stream>>>(W_lower, Wl_p, 256, 784, 256, 784, 256);
  dpc_pack_w<<<(800 * 64)  / 256, 256, 0, stream>>>(W_ih,    Wih_p, 800, 64, 784, 64, 784);
  dpc_pack_w<<<(64 * 64)   / 256, 256, 0, stream>>>(W_hh,    Whh_p, 64, 64, 64, 64, 64);
  dpc_pack_w<<<(64 * 128)  / 256, 256, 0, stream>>>(W_h1,    Wh1_p, 64, 128, 64, 128, 64);
  dpc_pack_w<<<(128 * 16)  / 256, 256, 0, stream>>>(W_h2,    Wh2_p, 128, 16, 128, 5, 128);
  for (int kk = 0; kk < 5; ++kk)
    dpc_pack_w<<<(256 * 256) / 256, 256, 0, stream>>>(Vk + (size_t)kk * 256 * 256,
                                                      Wv_p + (size_t)kk * 256 * 256,
                                                      256, 256, 256, 256, 256);

  dpc_step_kernel<<<NWG, NTHR, 0, stream>>>(
      inseq, noise, b_h1, b_h2, b_ih, b_hh, W_cls, b_cls,
      rt_bf, err_bf, rh_bf, h_bf, gw_f, vr_f,
      Wl_p, Wih_p, Whh_p, Wh1_p, Wh2_p, Wv_p,
      bar, pe_part, out);
}
